// SelfAttentionFCLayer_11149735101044
// MI455X (gfx1250) — matrix operand order repaired and checked by reading, not compiled, not run on hardware
//
#include <hip/hip_runtime.h>

#define BATCH 64
#define SEQ   1024
#define CDIM  128

typedef __attribute__((ext_vector_type(16))) _Float16 v16h;
typedef __attribute__((ext_vector_type(8)))  float    v8f;

static __device__ __forceinline__ v8f wmma_f16(v16h a, v16h b, v8f c) {
  return __builtin_amdgcn_wmma_f32_16x16x32_f16(false, a, false, b, (short)0, c,
                                                false, false);
}

// ---- operand loaders for CDNA5 WMMA (wave32); same contiguous k order in A and B ----
// A (16x32 f16): lane l holds row M=l&15; K=(l<16?0:16)+0..15 contiguous.
static __device__ __forceinline__ v16h load_a_f32(const float* __restrict__ row,
                                                  int cBase, int lane) {
  const float* p0 = row + cBase + ((lane < 16) ? 0 : 16);
  const float* p1 = p0 + 8;
  v16h r;
#pragma unroll
  for (int i = 0; i < 8; ++i) { r[i] = (_Float16)p0[i]; r[8 + i] = (_Float16)p1[i]; }
  return r;
}
static __device__ __forceinline__ v16h load_a_f16(const _Float16* __restrict__ row,
                                                  int cBase, int lane) {
  const _Float16* p = row + cBase + ((lane < 16) ? 0 : 16);
  v16h r;
#pragma unroll
  for (int i = 0; i < 8; ++i) { r[i] = p[i]; r[8 + i] = p[8 + i]; }
  return r;
}
// B (32x16 f16): lane l holds column N=l&15; K=(l<16?0:16)+0..15 contiguous.
// rowPtr must point at the source row feeding this lane's column.
static __device__ __forceinline__ v16h load_b_f32(const float* __restrict__ rowPtr,
                                                  int cBase, int lane) {
  const float* p = rowPtr + cBase + ((lane < 16) ? 0 : 16);
  v16h r;
#pragma unroll
  for (int i = 0; i < 16; ++i) r[i] = (_Float16)p[i];
  return r;
}
static __device__ __forceinline__ v16h load_b_f16(const _Float16* __restrict__ p) {
  return *(const v16h*)p;  // 16 contiguous halfs (32B)
}

// =====================================================================
// Kernel 1: Q = x W_q^T + b, K = x W_k^T + b (f16 row-major),
//           V^T[b][d][s] = (x W_v^T + b)^T  (f16, s-contiguous)
// =====================================================================
__global__ __launch_bounds__(256) void proj_kernel(
    const float* __restrict__ x,
    const float* __restrict__ qw, const float* __restrict__ qb,
    const float* __restrict__ kw, const float* __restrict__ kb,
    const float* __restrict__ vw, const float* __restrict__ vb,
    _Float16* __restrict__ Qh, _Float16* __restrict__ Kh,
    _Float16* __restrict__ Vt) {
  const int lane  = threadIdx.x & 31;
  const int wave  = threadIdx.x >> 5;
  const int mBase = blockIdx.x * 128 + wave * 16;
  const int colN  = lane & 15;
  const int hi8   = (lane >= 16) ? 8 : 0;

  // ---- Q and K: A = x rows (this wave's 16 rows), B = weight rows ----
  v16h Ax[4];
#pragma unroll
  for (int c = 0; c < 4; ++c)
    Ax[c] = load_a_f32(x + (size_t)(mBase + colN) * CDIM, c * 32, lane);

#pragma unroll
  for (int dt = 0; dt < 8; ++dt) {
    v8f accq = {}; v8f acck = {};
    const float* qwr = qw + (size_t)(dt * 16 + colN) * CDIM;
    const float* kwr = kw + (size_t)(dt * 16 + colN) * CDIM;
#pragma unroll
    for (int c = 0; c < 4; ++c) {
      accq = wmma_f16(Ax[c], load_b_f32(qwr, c * 32, lane), accq);
      acck = wmma_f16(Ax[c], load_b_f32(kwr, c * 32, lane), acck);
    }
    const float bq = qb[dt * 16 + colN];
    const float bk = kb[dt * 16 + colN];
#pragma unroll
    for (int r = 0; r < 8; ++r) {
      const size_t m = (size_t)(mBase + hi8 + r);
      Qh[m * CDIM + dt * 16 + colN] = (_Float16)(accq[r] + bq);
      Kh[m * CDIM + dt * 16 + colN] = (_Float16)(acck[r] + bk);
    }
  }

  // ---- V^T: A = W_v rows (d), B = x^T columns (this wave's 16 s) ----
  const int b     = mBase / SEQ;
  const int sBase = mBase % SEQ;
  v16h Bx[4];
#pragma unroll
  for (int c = 0; c < 4; ++c)
    Bx[c] = load_b_f32(x + (size_t)(mBase + colN) * CDIM, c * 32, lane);

#pragma unroll
  for (int dt = 0; dt < 8; ++dt) {
    v8f accv = {};
    const float* vwr = vw + (size_t)(dt * 16 + colN) * CDIM;
#pragma unroll
    for (int c = 0; c < 4; ++c)
      accv = wmma_f16(load_a_f32(vwr, c * 32, lane), Bx[c], accv);
#pragma unroll
    for (int r = 0; r < 8; ++r) {
      const int d = dt * 16 + hi8 + r;                   // M = d row
      Vt[((size_t)b * CDIM + d) * SEQ + sBase + colN] = (_Float16)(accv[r] + vb[d]);
    }
  }
}

// =====================================================================
// Kernel 2: flash attention. One 16-row q tile per wave.
// =====================================================================
__global__ __launch_bounds__(256) void attn_kernel(
    const _Float16* __restrict__ Qh, const _Float16* __restrict__ Kh,
    const _Float16* __restrict__ Vt, const int* __restrict__ vlen,
    float* __restrict__ out) {
  __shared__ _Float16 pbuf[8][16 * 32];  // per-wave P-tile transpose staging

  const int lane  = threadIdx.x & 31;
  const int wave  = threadIdx.x >> 5;
  const int bb    = blockIdx.x >> 3;                     // 8 blocks per batch
  const int qBase = ((blockIdx.x & 7) * 8 + wave) * 16;  // 64 q tiles per batch
  const int colN  = lane & 15;
  const int hi8   = (lane >= 16) ? 8 : 0;

  int vl = vlen[bb];
  if (vl < 0) vl = 0;
  if (vl > SEQ) vl = SEQ;

  float* outBase = out + (size_t)bb * SEQ * CDIM;

  if (qBase >= vl) {  // fully masked rows -> exact zeros
#pragma unroll
    for (int t = 0; t < 8; ++t)
#pragma unroll
      for (int r = 0; r < 8; ++r)
        outBase[(size_t)(qBase + hi8 + r) * CDIM + t * 16 + colN] = 0.0f;
    return;
  }

  // Q tile as 4 A-operands (d = 0..127 in chunks of 32)
  v16h Qa[4];
  const _Float16* qrow = Qh + (size_t)(bb * SEQ + qBase + colN) * CDIM;
#pragma unroll
  for (int c = 0; c < 4; ++c) Qa[c] = load_a_f16(qrow, c * 32, lane);

  const _Float16* Kbase = Kh + (size_t)bb * SEQ * CDIM;
  const _Float16* Vbase = Vt + (size_t)bb * CDIM * SEQ;

  v8f   O[8] = {};
  float mrow[8], lrow[8];
#pragma unroll
  for (int r = 0; r < 8; ++r) { mrow[r] = -3.0e38f; lrow[r] = 0.0f; }

  for (int kc = 0; kc < vl; kc += 32) {
    if (kc + 32 < vl) {  // CDNA5 global_prefetch for next chunk
      __builtin_prefetch(Kbase + (size_t)(kc + 32 + colN) * CDIM, 0, 1);
      __builtin_prefetch(Vbase + (size_t)colN * SEQ + kc + 32, 0, 1);
    }
    // scores: two 16x16 tiles (cols kc..+15, kc+16..+31), K=128 over 4 chunks
    v8f s0 = {}, s1 = {};
    const _Float16* kr0 = Kbase + (size_t)(kc + colN) * CDIM + ((lane < 16) ? 0 : 16);
#pragma unroll
    for (int c = 0; c < 4; ++c) s0 = wmma_f16(Qa[c], load_b_f16(kr0 + c * 32), s0);
    const _Float16* kr1 = kr0 + 16 * CDIM;
#pragma unroll
    for (int c = 0; c < 4; ++c) s1 = wmma_f16(Qa[c], load_b_f16(kr1 + c * 32), s1);

    const bool msk0 = (kc + colN) >= vl;
    const bool msk1 = (kc + 16 + colN) >= vl;
#pragma unroll
    for (int r = 0; r < 8; ++r) {
      if (msk0) s0[r] = -3.0e38f;
      if (msk1) s1[r] = -3.0e38f;
    }
    // per-row max across the 16-lane half (rows r / r+8 live in lanes 0-15 / 16-31)
    float cmax[8];
#pragma unroll
    for (int r = 0; r < 8; ++r) cmax[r] = fmaxf(s0[r], s1[r]);
#pragma unroll
    for (int off = 1; off < 16; off <<= 1)
#pragma unroll
      for (int r = 0; r < 8; ++r) cmax[r] = fmaxf(cmax[r], __shfl_xor(cmax[r], off));

    float alpha[8], rsum[8];
#pragma unroll
    for (int r = 0; r < 8; ++r) {
      const float mnew = fmaxf(mrow[r], cmax[r]);
      alpha[r] = __expf(mrow[r] - mnew);
      mrow[r]  = mnew;
      const float p0 = msk0 ? 0.0f : __expf(s0[r] - mnew);
      const float p1 = msk1 ? 0.0f : __expf(s1[r] - mnew);
      rsum[r] = p0 + p1;
      pbuf[wave][(r + hi8) * 32 + colN]      = (_Float16)p0;
      pbuf[wave][(r + hi8) * 32 + colN + 16] = (_Float16)p1;
    }
#pragma unroll
    for (int off = 1; off < 16; off <<= 1)
#pragma unroll
      for (int r = 0; r < 8; ++r) rsum[r] += __shfl_xor(rsum[r], off);
#pragma unroll
    for (int r = 0; r < 8; ++r) lrow[r] = lrow[r] * alpha[r] + rsum[r];
#pragma unroll
    for (int t = 0; t < 8; ++t)
#pragma unroll
      for (int r = 0; r < 8; ++r) O[t][r] *= alpha[r];

    // read P back in A layout (C->A transpose via LDS, same wave, in-order DS)
    const _Float16* prow = &pbuf[wave][colN * 32 + ((lane < 16) ? 0 : 16)];
    v16h Pa;
#pragma unroll
    for (int i = 0; i < 8; ++i) { Pa[i] = prow[i]; Pa[8 + i] = prow[8 + i]; }

    // O += P (16x32 over k) x V^T (B: column = d, K = k contiguous)
#pragma unroll
    for (int t = 0; t < 8; ++t) {
      const _Float16* vr =
          Vbase + (size_t)(t * 16 + colN) * SEQ + kc + ((lane < 16) ? 0 : 16);
      O[t] = wmma_f16(Pa, load_b_f16(vr), O[t]);
    }
  }

  // epilogue: normalize; rows q >= vl are exactly zero (matches attn*(1-mask))
  float rinv[8];
#pragma unroll
  for (int r = 0; r < 8; ++r) rinv[r] = (lrow[r] > 0.0f) ? 1.0f / lrow[r] : 0.0f;
#pragma unroll
  for (int t = 0; t < 8; ++t)
#pragma unroll
    for (int r = 0; r < 8; ++r) {
      const int q = qBase + hi8 + r;
      outBase[(size_t)q * CDIM + t * 16 + colN] = (q < vl) ? O[t][r] * rinv[r] : 0.0f;
    }
}

extern "C" void kernel_launch(void* const* d_in, const int* in_sizes, int n_in,
                              void* d_out, int out_size, void* d_ws, size_t ws_size,
                              hipStream_t stream) {
  (void)in_sizes; (void)n_in; (void)out_size; (void)ws_size;
  const float* x  = (const float*)d_in[0];
  const float* qw = (const float*)d_in[1];
  const float* qb = (const float*)d_in[2];
  const float* kw = (const float*)d_in[3];
  const float* kb = (const float*)d_in[4];
  const float* vw = (const float*)d_in[5];
  const float* vb = (const float*)d_in[6];
  const int*   vl = (const int*)d_in[7];
  float* out = (float*)d_out;

  char* ws = (char*)d_ws;
  const size_t matBytes = (size_t)BATCH * SEQ * CDIM * sizeof(_Float16);  // 16 MB
  _Float16* Qh = (_Float16*)(ws);
  _Float16* Kh = (_Float16*)(ws + matBytes);
  _Float16* Vt = (_Float16*)(ws + 2 * matBytes);

  proj_kernel<<<(BATCH * SEQ) / 128, 256, 0, stream>>>(x, qw, qb, kw, kb, vw, vb,
                                                       Qh, Kh, Vt);
  attn_kernel<<<BATCH * 8, 256, 0, stream>>>(Qh, Kh, Vt, vl, out);
}
